// APPNP_38216619000378
// MI455X (gfx1250) — matrix operand order repaired and checked by reading, not compiled, not run on hardware
//
#include <hip/hip_runtime.h>

#define N_NODES_C 100000
#define D_IN_C    256
#define D_HID_C   64
#define D_OUT_C   32
#define NUM_PROP_C 10
#define ALPHA_C   0.1f

#define HID_STRIDE 72   // padded row stride (halfs) -> 144B, bank-conflict free

typedef __attribute__((ext_vector_type(16))) _Float16 v16h;
typedef __attribute__((ext_vector_type(8)))  _Float16 v8h;
typedef __attribute__((ext_vector_type(8)))  float    v8f;

// ---------------------------------------------------------------------------
// Kernel 1: fused 2-layer MLP head via f16 WMMA, fp32 accumulate.
// One wave32 computes one 16-row node tile. 8 waves / block.
// W1/W2 are pre-swizzled into WMMA B-fragment order in LDS so every B operand
// is one aligned 32-byte LDS read (2x ds_load_b128), lane-linear addressing.
// ---------------------------------------------------------------------------
__global__ __launch_bounds__(256) void appnp_mlp_wmma(
    const float* __restrict__ feat,  // [N, 256]
    const float* __restrict__ W1,    // [256, 64]
    const float* __restrict__ b1,    // [64]
    const float* __restrict__ W2,    // [64, 32]
    const float* __restrict__ b2,    // [32]
    float* __restrict__ x0,          // [N, 32] out
    float* __restrict__ x)           // [N, 32] out (current state)
{
    // B fragments: [chunk][ntile][lane][16 elems]
    __shared__ alignas(32) _Float16 sW1B[8 * 4 * 32 * 16];   // 32 KB
    __shared__ alignas(32) _Float16 sW2B[2 * 2 * 32 * 16];   //  4 KB
    __shared__ alignas(16) _Float16 sHid[8][16 * HID_STRIDE];// 18 KB

    // ---- One-time fragment-order staging (all 256 threads, then sync) ----
    // B layout (16x16x32 f16): lane holds N = lane&15; elem e -> K = (lane>>4)*8 + (e<8 ? e : 8+e)
    for (int p = threadIdx.x; p < 8 * 4 * 32 * 16; p += 256) {
        const int e = p & 15;
        const int l = (p >> 4) & 31;
        const int t = (p >> 9) & 3;
        const int c = p >> 11;
        const int k = c * 32 + (l >> 4) * 8 + (e < 8 ? e : 8 + e);
        const int n = t * 16 + (l & 15);
        sW1B[p] = (_Float16)W1[k * D_HID_C + n];
    }
    for (int p = threadIdx.x; p < 2 * 2 * 32 * 16; p += 256) {
        const int e = p & 15;
        const int l = (p >> 4) & 31;
        const int t = (p >> 9) & 1;
        const int c = p >> 10;
        const int k = c * 32 + (l >> 4) * 8 + (e < 8 ? e : 8 + e);
        const int n = t * 16 + (l & 15);
        sW2B[p] = (_Float16)W2[k * D_OUT_C + n];
    }
    __syncthreads();

    const int lane = threadIdx.x & 31;
    const int wave = threadIdx.x >> 5;
    const int tile = blockIdx.x * 8 + wave;        // 16-row tile index
    if (tile >= N_NODES_C / 16) return;            // no further block-wide sync

    const int m0   = tile * 16;
    const int half = lane >> 4;                    // 0 | 1 (K half selector)
    const int l15  = lane & 15;

    // ---- GEMM1: hid[16x64] = relu(feat_tile[16x256] @ W1 + b1) ----
    v8f acc[4];
    #pragma unroll
    for (int t = 0; t < 4; ++t) acc[t] = (v8f){};

    const float* arow = feat + (size_t)(m0 + l15) * D_IN_C;
    #pragma unroll
    for (int c = 0; c < 8; ++c) {
        // A 16x32 f16: lane = M, elem e -> K = c*32 + half*8 + (e<8 ? e : 16+e-8)
        v16h a;
        const float* ap = arow + c * 32 + half * 8;
        #pragma unroll
        for (int e = 0; e < 8; ++e) a[e]     = (_Float16)ap[e];
        #pragma unroll
        for (int e = 0; e < 8; ++e) a[8 + e] = (_Float16)ap[16 + e];

        #pragma unroll
        for (int t = 0; t < 4; ++t) {
            const v16h b = *(const v16h*)&sW1B[((c * 4 + t) * 32 + lane) * 16];
            acc[t] = __builtin_amdgcn_wmma_f32_16x16x32_f16(
                false, a, false, b, (short)0, acc[t], false, false);
        }
    }

    // bias + relu -> LDS row-major (re-stripes C/D layout into A layout source)
    _Float16* hp = sHid[wave];
    #pragma unroll
    for (int t = 0; t < 4; ++t) {
        const int n = t * 16 + l15;
        const float bias = b1[n];
        #pragma unroll
        for (int v = 0; v < 8; ++v) {
            const int m = v + half * 8;            // C/D: VGPR v, lane-half -> M
            float h = acc[t][v] + bias;
            hp[m * HID_STRIDE + n] = (_Float16)(h > 0.0f ? h : 0.0f);
        }
    }

    // ---- GEMM2: out[16x32] = hid[16x64] @ W2 + b2 ----
    v8f o[2];
    o[0] = (v8f){}; o[1] = (v8f){};
    #pragma unroll
    for (int c = 0; c < 2; ++c) {
        const _Float16* ap = hp + l15 * HID_STRIDE + c * 32 + half * 8;
        const v8h lo = *(const v8h*)ap;            // 16B aligned (144|16, 64|16)
        const v8h hi = *(const v8h*)(ap + 16);
        const v16h a = __builtin_shufflevector(lo, hi,
            0, 1, 2, 3, 4, 5, 6, 7, 8, 9, 10, 11, 12, 13, 14, 15);

        #pragma unroll
        for (int t = 0; t < 2; ++t) {
            const v16h b = *(const v16h*)&sW2B[((c * 2 + t) * 32 + lane) * 16];
            o[t] = __builtin_amdgcn_wmma_f32_16x16x32_f16(
                false, a, false, b, (short)0, o[t], false, false);
        }
    }

    #pragma unroll
    for (int t = 0; t < 2; ++t) {
        const int n = t * 16 + l15;
        const float bias = b2[n];
        #pragma unroll
        for (int v = 0; v < 8; ++v) {
            const int m = v + half * 8;
            const float r = o[t][v] + bias;
            const size_t idx = (size_t)(m0 + m) * D_OUT_C + n;
            x0[idx] = r;
            x[idx]  = r;
        }
    }
}

// ---------------------------------------------------------------------------
// Kernel 2: dst = alpha * x0   (vectorized float4, initializes accumulator)
// ---------------------------------------------------------------------------
__global__ __launch_bounds__(256) void appnp_scale(
    const float4* __restrict__ x0, float4* __restrict__ y, int n4)
{
    const int i = blockIdx.x * 256 + threadIdx.x;
    if (i < n4) {
        float4 v = x0[i];
        float4 r;
        r.x = ALPHA_C * v.x; r.y = ALPHA_C * v.y;
        r.z = ALPHA_C * v.z; r.w = ALPHA_C * v.w;
        y[i] = r;
    }
}

// ---------------------------------------------------------------------------
// Kernel 3: COO SpMM scatter. One edge per wave32: lane == channel (D_OUT=32).
// Gather x[col[e]] (coalesced 128B) and global_atomic_add_f32 into y[row[e]].
// ---------------------------------------------------------------------------
__global__ __launch_bounds__(256) void appnp_spmm(
    const int* __restrict__ row, const int* __restrict__ col,
    const float* __restrict__ vals,
    const float* __restrict__ xin, float* __restrict__ y, int nE)
{
    const int e = blockIdx.x * 8 + (int)(threadIdx.x >> 5);
    if (e >= nE) return;
    const int lane = threadIdx.x & 31;
    const int r = row[e];
    const int c = col[e];
    const float w = (1.0f - ALPHA_C) * vals[e];
    const float v = xin[(size_t)c * D_OUT_C + lane];
    unsafeAtomicAdd(&y[(size_t)r * D_OUT_C + lane], w * v);
}

// ---------------------------------------------------------------------------
extern "C" void kernel_launch(void* const* d_in, const int* in_sizes, int n_in,
                              void* d_out, int out_size, void* d_ws, size_t ws_size,
                              hipStream_t stream)
{
    const float* feat = (const float*)d_in[0];
    const int*   row  = (const int*)d_in[1];
    const int*   col  = (const int*)d_in[2];
    const float* vals = (const float*)d_in[3];
    const float* W1   = (const float*)d_in[4];
    const float* b1   = (const float*)d_in[5];
    const float* W2   = (const float*)d_in[6];
    const float* b2   = (const float*)d_in[7];

    float* xout = (float*)d_out;                         // current state / final out
    float* x0   = (float*)d_ws;                          // 12.8 MB
    float* xb   = x0 + (size_t)N_NODES_C * D_OUT_C;      // 12.8 MB ping buffer
    const int nE = in_sizes[1];

    // MLP head: 6250 row-tiles, 8 tiles (waves) per block.
    const int nTiles = N_NODES_C / 16;
    appnp_mlp_wmma<<<dim3((nTiles + 7) / 8), 256, 0, stream>>>(
        feat, W1, b1, W2, b2, x0, xout);

    const int n4 = (N_NODES_C * D_OUT_C) / 4;
    for (int it = 0; it < NUM_PROP_C; ++it) {
        float* src = (it & 1) ? xb : xout;               // it=0 reads d_out
        float* dst = (it & 1) ? xout : xb;               // it=9 writes d_out
        appnp_scale<<<dim3((n4 + 255) / 256), 256, 0, stream>>>(
            (const float4*)x0, (float4*)dst, n4);
        appnp_spmm<<<dim3((nE + 7) / 8), 256, 0, stream>>>(
            row, col, vals, src, dst, nE);
    }
}
